// GeneNetwork_12747462934610
// MI455X (gfx1250) — compile-verified, matrix operand order checked by reading, not run on hardware
//
#include <hip/hip_runtime.h>
#include <math.h>

// GeneNetwork TAGConv forward for gfx1250 (MI455X).
// Dense matmuls -> v_wmma_f32_16x16x32_bf16: one wave computes a 16x64 output
// strip (4 col-tiles share one converted A fragment); weights pre-swizzled to
// the WMMA B-register layout once per launch (b128 loads, no conversion).
// Sparse hops -> CSR-by-destination built once per launch, then each hop is an
// atomic-free gather segment-sum (one wave per node) -- removes ~614M f32
// atomics/launch, which would otherwise dominate (atomic-op rate >> HBM time).
// Assumes N (node count) is a multiple of 16 (100000 = 6250*16), N <= 262144.

#define N_IN 96
#define DD   64
#define SCAN_B 256

typedef __attribute__((ext_vector_type(16))) __bf16        v16bf;
typedef __attribute__((ext_vector_type(8)))  float         v8f;
typedef __attribute__((ext_vector_type(8)))  unsigned int  v8u;

__device__ __forceinline__ unsigned int f2bf1(float f) {
  unsigned int u = __float_as_uint(f);
  return (u + 0x7FFFu + ((u >> 16) & 1u)) >> 16;   // round-to-nearest-even
}
__device__ __forceinline__ unsigned int packbf(float lo, float hi) {
#if __has_builtin(__builtin_amdgcn_cvt_pk_bf16_f32)
  return __builtin_bit_cast(unsigned int,
                            __builtin_amdgcn_cvt_pk_bf16_f32(lo, hi));
#else
  return f2bf1(lo) | (f2bf1(hi) << 16);
#endif
}

// ---------------------------------------------------------------------------
// Weight swizzle: B[K,64] fp32 row-major -> bf16 in WMMA B-register order.
// uint index u = ((ct*KSTEPS + ks)*32 + lane)*8 + j2 ; uint j2 holds
// elements {2*j2, 2*j2+1} of the lane's v16bf.
// ---------------------------------------------------------------------------
__global__ void convert_wsw(const float* __restrict__ B,
                            unsigned int* __restrict__ out, int Ksteps)
{
  int u = blockIdx.x * blockDim.x + threadIdx.x;
  if (u >= 4 * Ksteps * 32 * 8) return;
  int j2   = u & 7;
  int lane = (u >> 3) & 31;
  int t    = u >> 8;              // ct*Ksteps + ks
  int ks   = t % Ksteps;
  int ct   = t / Ksteps;
  int kb   = (lane >> 4) * 8;
  int col  = ct * 16 + (lane & 15);
  int k0   = ks * 32 + kb + (j2 < 4 ? 2 * j2 : 16 + 2 * (j2 - 4));
  out[u] = packbf(B[k0 * DD + col], B[(k0 + 1) * DD + col]);
}

// ---------------------------------------------------------------------------
// WMMA GEMM: Out[N,64] (=|+=) A[N,KDIM] @ B[KDIM,64] (+bias,ELU for pre_mp).
// One wave per 16-row block; converted A fragment feeds 4 column tiles.
// ---------------------------------------------------------------------------
template <int KDIM, bool ACCUM, bool BIASELU>
__global__ void __launch_bounds__(32)
gemm_wmma(const float* __restrict__ A, const unsigned int* __restrict__ Bsw,
          const float* __restrict__ bias, float* __restrict__ Out)
{
  constexpr int KSTEPS = KDIM / 32;
  const int lane    = threadIdx.x & 31;
  const int rowbase = blockIdx.x * 16;
  const int m  = lane & 15;
  const int hi = lane >> 4;
  const int kb = hi * 8;

  v8f c[4];
#pragma unroll
  for (int ct = 0; ct < 4; ++ct) {
    if constexpr (ACCUM) {
#pragma unroll
      for (int v = 0; v < 8; ++v)
        c[ct][v] = Out[(size_t)(rowbase + v + 8 * hi) * DD + ct * 16 + m];
    } else {
      c[ct] = (v8f){};
    }
  }

  const float* Ap = A + (size_t)(rowbase + m) * KDIM + kb;

#pragma unroll
  for (int ks = 0; ks < KSTEPS; ++ks) {
    const float* a0 = Ap + ks * 32;
    v8u au;
#pragma unroll
    for (int j = 0; j < 4; ++j) {
      au[j]     = packbf(a0[2 * j],      a0[2 * j + 1]);
      au[4 + j] = packbf(a0[16 + 2 * j], a0[17 + 2 * j]);
    }
    v16bf av = __builtin_bit_cast(v16bf, au);
#pragma unroll
    for (int ct = 0; ct < 4; ++ct) {
      const uint4* bp =
          (const uint4*)Bsw + (((size_t)ct * KSTEPS + ks) * 32 + lane) * 2;
      uint4 b0 = bp[0];
      uint4 b1 = bp[1];
      v8u bu;
      bu[0] = b0.x; bu[1] = b0.y; bu[2] = b0.z; bu[3] = b0.w;
      bu[4] = b1.x; bu[5] = b1.y; bu[6] = b1.z; bu[7] = b1.w;
      v16bf bv = __builtin_bit_cast(v16bf, bu);
      c[ct] = __builtin_amdgcn_wmma_f32_16x16x32_bf16(false, av, false, bv,
                                                      (short)0, c[ct],
                                                      false, false);
    }
  }

#pragma unroll
  for (int ct = 0; ct < 4; ++ct) {
    const int col = ct * 16 + m;
#pragma unroll
    for (int v = 0; v < 8; ++v) {
      float x = c[ct][v];
      if constexpr (BIASELU) {
        x += bias[col];
        x = x > 0.f ? x : (__expf(x) - 1.f);
      }
      Out[(size_t)(rowbase + v + 8 * hi) * DD + col] = x;
    }
  }
}

// ---------------------------------------------------------------------------
// CSR construction (once per launch)
// ---------------------------------------------------------------------------
__global__ void fill_zero(float* __restrict__ p, size_t n) {
  size_t i = (size_t)blockIdx.x * blockDim.x + threadIdx.x;
  if (i < n) p[i] = 0.f;
}

__global__ void deg_count(const int* __restrict__ dst, int* __restrict__ cnt, int E) {
  int e = blockIdx.x * blockDim.x + threadIdx.x;
  if (e < E) atomicAdd(&cnt[dst[e]], 1);
}

__global__ void make_dis(const int* __restrict__ cnt, float* __restrict__ dis, int N) {
  int i = blockIdx.x * blockDim.x + threadIdx.x;
  if (i < N) {
    int c = cnt[i];
    dis[i] = c > 0 ? rsqrtf((float)c) : 0.f;
  }
}

// Exclusive scan, 3 phases. nb = ceil(N/SCAN_B) must be <= 1024.
__global__ void scan_block(const int* __restrict__ in, int* __restrict__ out,
                           int* __restrict__ sums, int n)
{
  __shared__ int tmp[SCAN_B];
  int gid = blockIdx.x * SCAN_B + threadIdx.x;
  int v = (gid < n) ? in[gid] : 0;
  tmp[threadIdx.x] = v;
  __syncthreads();
  for (int off = 1; off < SCAN_B; off <<= 1) {
    int t = (threadIdx.x >= off) ? tmp[threadIdx.x - off] : 0;
    __syncthreads();
    tmp[threadIdx.x] += t;
    __syncthreads();
  }
  if (gid < n) out[gid] = tmp[threadIdx.x] - v;            // exclusive
  if (threadIdx.x == SCAN_B - 1) sums[blockIdx.x] = tmp[threadIdx.x];
}

__global__ void scan_aux(int* __restrict__ sums, int nb) {
  __shared__ int tmp[1024];
  int t = threadIdx.x;
  int v = (t < nb) ? sums[t] : 0;
  tmp[t] = v;
  __syncthreads();
  for (int off = 1; off < 1024; off <<= 1) {
    int u = (t >= off) ? tmp[t - off] : 0;
    __syncthreads();
    tmp[t] += u;
    __syncthreads();
  }
  if (t < nb) sums[t] = tmp[t] - v;                        // exclusive
}

__global__ void scan_add(int* __restrict__ out, const int* __restrict__ sums, int n) {
  int gid = blockIdx.x * SCAN_B + threadIdx.x;
  if (gid < n) out[gid] += sums[blockIdx.x];
}

__global__ void set_int(int* __restrict__ p, int v) { p[0] = v; }

__global__ void edge_place(const int* __restrict__ src, const int* __restrict__ dst,
                           const float* __restrict__ dis, int* __restrict__ cursor,
                           int* __restrict__ csr_src, float* __restrict__ csr_norm,
                           int E)
{
  int e = blockIdx.x * blockDim.x + threadIdx.x;
  if (e >= E) return;
  int s = src[e], d = dst[e];
  int pos = atomicAdd(&cursor[d], 1);
  csr_src[pos] = s;
  csr_norm[pos] = dis[s] * dis[d];
}

// ---------------------------------------------------------------------------
// Atomic-free hop: out[n,:] = sum_{e: dst==n} norm_e * in[src_e,:]
// One wave per node; lane accumulates 2 features in registers.
// ---------------------------------------------------------------------------
__global__ void __launch_bounds__(256)
gather_hop(const int* __restrict__ rowstart, const int* __restrict__ csr_src,
           const float* __restrict__ csr_norm,
           const float* __restrict__ in, float* __restrict__ out, int N)
{
  int node = blockIdx.x * 8 + (threadIdx.x >> 5);
  int lane = threadIdx.x & 31;
  if (node >= N) return;
  int f = lane * 2;
  float a0 = 0.f, a1 = 0.f;
  int p1 = rowstart[node + 1];
  for (int p = rowstart[node]; p < p1; ++p) {
    int s = csr_src[p];
    float nrm = csr_norm[p];
    const float2 v = *(const float2*)(in + (size_t)s * DD + f);
    a0 += nrm * v.x;
    a1 += nrm * v.y;
  }
  float* o = out + (size_t)node * DD + f;
  o[0] = a0;
  o[1] = a1;
}

// ---------------------------------------------------------------------------
// bias + ELU + LayerNorm(feature dim), one wave per node (2 feats/lane).
__global__ void __launch_bounds__(256)
elu_ln(const float* __restrict__ O, const float* __restrict__ tb,
       const float* __restrict__ g, const float* __restrict__ bb,
       float* __restrict__ H, int N)
{
  int node = blockIdx.x * 8 + (threadIdx.x >> 5);
  int lane = threadIdx.x & 31;
  if (node >= N) return;
  int f = lane * 2;
  const float* row = O + (size_t)node * DD;
  float v0 = row[f]     + tb[f];
  float v1 = row[f + 1] + tb[f + 1];
  v0 = v0 > 0.f ? v0 : (__expf(v0) - 1.f);
  v1 = v1 > 0.f ? v1 : (__expf(v1) - 1.f);
  float s  = v0 + v1;
  float s2 = v0 * v0 + v1 * v1;
  for (int off = 16; off > 0; off >>= 1) {
    s  += __shfl_xor(s,  off, 32);
    s2 += __shfl_xor(s2, off, 32);
  }
  float mu  = s * (1.f / 64.f);
  float var = s2 * (1.f / 64.f) - mu * mu;
  float inv = rsqrtf(var + 1e-5f);
  float* hrow = H + (size_t)node * DD;
  hrow[f]     = (v0 - mu) * inv * g[f]     + bb[f];
  hrow[f + 1] = (v1 - mu) * inv * g[f + 1] + bb[f + 1];
}

// out[node] = dot(h[node,:], post_W) + post_b; one wave per node.
__global__ void __launch_bounds__(256)
post_dot(const float* __restrict__ H, const float* __restrict__ W,
         const float* __restrict__ b, float* __restrict__ out, int N)
{
  int node = blockIdx.x * 8 + (threadIdx.x >> 5);
  int lane = threadIdx.x & 31;
  if (node >= N) return;
  const float* row = H + (size_t)node * DD;
  float s = row[lane * 2] * W[lane * 2] + row[lane * 2 + 1] * W[lane * 2 + 1];
  for (int off = 16; off > 0; off >>= 1) s += __shfl_xor(s, off, 32);
  if (lane == 0) out[node] = s + b[0];
}

// ---------------------------------------------------------------------------
extern "C" void kernel_launch(void* const* d_in, const int* in_sizes, int n_in,
                              void* d_out, int out_size, void* d_ws, size_t ws_size,
                              hipStream_t stream)
{
  const float* x      = (const float*)d_in[0];
  const int*   ei     = (const int*)  d_in[1];
  const float* pre_W  = (const float*)d_in[2];
  const float* pre_b  = (const float*)d_in[3];
  const float* tag_W  = (const float*)d_in[4];   // [L,4,64,64]
  const float* tag_b  = (const float*)d_in[5];   // [L,64]
  const float* ln_g   = (const float*)d_in[6];
  const float* ln_b   = (const float*)d_in[7];
  const float* post_W = (const float*)d_in[8];
  const float* post_b = (const float*)d_in[9];

  const int N = in_sizes[0] / N_IN;
  const int E = in_sizes[1] / 2;
  const int* srcI = ei;
  const int* dstI = ei + E;

  const int PRE_U = 4 * 3 * 32 * 8;   // pre_W swizzled uints (K=96 -> 3 ksteps)
  const int TAG_U = 4 * 2 * 32 * 8;   // per tag matrix (K=64 -> 2 ksteps)
  const int nb    = (N + SCAN_B - 1) / SCAN_B;

  // Workspace layout (~116 MB):
  // h | out | bufA | bufB (N*64 f32 each, 16B aligned) | wsw (bf16 weights)
  // | csr_norm (E f32) | csr_src (E int) | dis (N) | rowstart (N+1)
  // | counts (N) | cursor (N) | aux (nb)
  float* h    = (float*)d_ws;
  float* out  = h    + (size_t)N * DD;
  float* bufA = out  + (size_t)N * DD;
  float* bufB = bufA + (size_t)N * DD;
  unsigned int* wsw = (unsigned int*)(bufB + (size_t)N * DD);
  unsigned int* preSW = wsw;
  unsigned int* tagSW = wsw + PRE_U;
  float* csr_norm = (float*)(tagSW + 8 * TAG_U);
  int*   csr_src  = (int*)(csr_norm + E);
  float* dis      = (float*)(csr_src + E);
  int*   rowstart = (int*)(dis + N);
  int*   counts   = rowstart + (N + 1);
  int*   cursor   = counts + N;
  int*   aux      = cursor + N;

  const int gemmGrid = N / 16;
  const int nodeGrid = (N + 7) / 8;

  // Stage weights as bf16 in WMMA B-register layout (tiny, once per launch)
  convert_wsw<<<(PRE_U + 255) / 256, 256, 0, stream>>>(pre_W, preSW, 3);
  for (int m = 0; m < 8; ++m)
    convert_wsw<<<(TAG_U + 255) / 256, 256, 0, stream>>>(
        tag_W + (size_t)m * DD * DD, tagSW + (size_t)m * TAG_U, 2);

  // ---- CSR build: counts -> dis, exclusive scan -> rowstart, place edges
  fill_zero<<<(N + 255) / 256, 256, 0, stream>>>((float*)counts, (size_t)N);
  deg_count<<<(E + 255) / 256, 256, 0, stream>>>(dstI, counts, E);
  make_dis <<<(N + 255) / 256, 256, 0, stream>>>(counts, dis, N);
  scan_block<<<nb, SCAN_B, 0, stream>>>(counts, rowstart, aux, N);
  scan_aux  <<<1, 1024, 0, stream>>>(aux, nb);
  scan_add  <<<nb, SCAN_B, 0, stream>>>(rowstart, aux, N);
  set_int   <<<1, 1, 0, stream>>>(rowstart + N, E);
  hipMemcpyAsync(cursor, rowstart, (size_t)N * sizeof(int),
                 hipMemcpyDeviceToDevice, stream);
  edge_place<<<(E + 255) / 256, 256, 0, stream>>>(srcI, dstI, dis, cursor,
                                                  csr_src, csr_norm, E);

  // pre_mp: h = ELU(x @ pre_W + pre_b)
  gemm_wmma<N_IN, false, true><<<gemmGrid, 32, 0, stream>>>(x, preSW, pre_b, h);

  for (int l = 0; l < 2; ++l) {
    const unsigned int* Wl = tagSW + (size_t)l * 4 * TAG_U;
    // out = h @ W_0
    gemm_wmma<DD, false, false><<<gemmGrid, 32, 0, stream>>>(h, Wl, nullptr, out);
    const float* srcf = h;
    float* dstf = bufB;
    float* other = bufA;
    for (int k = 1; k <= 3; ++k) {
      gather_hop<<<nodeGrid, 256, 0, stream>>>(rowstart, csr_src, csr_norm,
                                               srcf, dstf, N);
      // out += hk @ W_k
      gemm_wmma<DD, true, false><<<gemmGrid, 32, 0, stream>>>(
          dstf, Wl + (size_t)k * TAG_U, nullptr, out);
      srcf = dstf;
      float* t = dstf; dstf = other; other = t;
    }
    // h = LayerNorm(ELU(out + tag_b[l]))
    elu_ln<<<nodeGrid, 256, 0, stream>>>(out, tag_b + l * DD,
                                         ln_g + l * DD, ln_b + l * DD, h, N);
  }

  // post_mp
  post_dot<<<nodeGrid, 256, 0, stream>>>(h, post_W, post_b, (float*)d_out, N);
}
